// PyramidCrossAttention_6966436954356
// MI455X (gfx1250) — matrix operand order repaired and checked by reading, not compiled, not run on hardware
//
#include <hip/hip_runtime.h>
#include <math.h>

// ---------------------------------------------------------------------------
// PyramidCrossAttention for MI455X (gfx1250, wave32, WMMA).
// GEMMs: bf16 WMMA, 2x2 register blocking per wave (4 v_wmma / k-step),
// double-buffered LDS staging (1 barrier / k-step), fused epilogues,
// native f32->bf16 converts, packed b32 transposed LDS stores.
// ---------------------------------------------------------------------------

typedef __attribute__((ext_vector_type(16))) __bf16 v16bf;
typedef __attribute__((ext_vector_type(8)))  __bf16 v8bf;
typedef __attribute__((ext_vector_type(8)))  float  v8f;

union FragV { v16bf v; v8bf h[2]; };

static __device__ __forceinline__ unsigned short bf16bits(float f) {
    return __builtin_bit_cast(unsigned short, (__bf16)f);
}

// Problem constants
#define BB 4
#define CC 256
#define HH 64
#define WW 64
#define QQ 4096
#define LL 3
#define NH 8
#define NP 4
#define DD 32
#define CM 128
#define GG 8

// ---------------------------------------------------------------------------
// 3x3 box blur (zero pad, /9) -> q_low, q_high
// ---------------------------------------------------------------------------
__global__ __launch_bounds__(256) void blur_kernel(
    const float* __restrict__ q, float* __restrict__ qlow,
    float* __restrict__ qhigh, int total)
{
    int idx = blockIdx.x * 256 + threadIdx.x;
    if (idx >= total) return;
    int x = idx & 63;
    int y = (idx >> 6) & 63;
    const float* base = q + (idx - (y * 64 + x));
    float s = 0.f;
    for (int dy = -1; dy <= 1; ++dy) {
        int yy = y + dy; if (yy < 0 || yy > 63) continue;
        for (int dx = -1; dx <= 1; ++dx) {
            int xx = x + dx; if (xx < 0 || xx > 63) continue;
            s += base[yy * 64 + xx];
        }
    }
    float low = s * (1.f / 9.f);
    qlow[idx]  = low;
    qhigh[idx] = q[idx] - low;
}

// ---------------------------------------------------------------------------
// GroupNorm statistics: one block per (batch, group); group = cpg contiguous
// channels of [b][c][q] layout -> contiguous cpg*Q floats.
// stat layout: [b][g][2] = {mean, rstd}
// ---------------------------------------------------------------------------
__global__ __launch_bounds__(256) void gn_stats_kernel(
    const float* __restrict__ x, float* __restrict__ stat,
    int CMv, int Qv, int cpg)
{
    int g = blockIdx.x, b = blockIdx.y;
    size_t base = ((size_t)b * CMv + (size_t)g * cpg) * Qv;
    int n = cpg * Qv;
    float s = 0.f, sq = 0.f;
    for (int i = threadIdx.x; i < n; i += 256) {
        float v = x[base + i]; s += v; sq += v * v;
    }
    __shared__ float ls[256], lq[256];
    ls[threadIdx.x] = s; lq[threadIdx.x] = sq;
    __syncthreads();
    for (int st = 128; st > 0; st >>= 1) {
        if (threadIdx.x < st) {
            ls[threadIdx.x] += ls[threadIdx.x + st];
            lq[threadIdx.x] += lq[threadIdx.x + st];
        }
        __syncthreads();
    }
    if (threadIdx.x == 0) {
        float mean = ls[0] / n;
        float var  = lq[0] / n - mean * mean;
        stat[(b * GG + g) * 2 + 0] = mean;
        stat[(b * GG + g) * 2 + 1] = rsqrtf(var + 1e-5f);
    }
}

// ---------------------------------------------------------------------------
// Batched WMMA GEMM:  C[z] = A(MxK) * Bsrc[z](KxN), bf16 inputs, f32 accum.
// Block = 256 threads = 8 waves (2 mw x 4 nw), block tile 64(M) x 128(N),
// wave tile 32x32 (2x2 fragments -> 4 v_wmma per 32-wide k-step).
// Fused on B:  + kadd[(z%kadd_mod)*K + k]  (level embed)
//              GroupNorm (gnstat/gamma/beta/cpg) + optional ReLU
// Fused on C:  + bias[m], + addfull[z,m,n]*(*scale_ptr), optional transpose-out.
// ---------------------------------------------------------------------------
__global__ __launch_bounds__(256) void wmma_gemm_bf16(
    const float* __restrict__ A, const float* __restrict__ Bm,
    float* __restrict__ Cm,
    int M, int N, int K,
    const float* __restrict__ bias,
    const float* __restrict__ kadd, int kadd_mod,
    const float* __restrict__ gnstat, const float* __restrict__ gamma,
    const float* __restrict__ beta, int cpg, int relu,
    const float* __restrict__ addfull, const float* __restrict__ scale_ptr,
    int out_transpose)
{
    constexpr int LDK = 40;  // 80B pitch: 16B-aligned fragments, staggered banks
    __shared__ __bf16 ldsA[2][64 * LDK];
    __shared__ __bf16 ldsB[2][128 * LDK];

    const int t     = threadIdx.x;
    const int z     = blockIdx.z;
    const int mBase = blockIdx.y * 64;
    const int nBase = blockIdx.x * 128;
    const int nk    = K >> 5;

    const int wid  = t >> 5;
    const int lane = t & 31;
    const int hf   = lane >> 4;   // lane-half selects K sub-block
    const int r16  = lane & 15;   // row (A:m, B:n) within 16-tile
    const int mw   = wid >> 2;    // wave's 32-row M tile (0..1)
    const int nw   = wid & 3;     // wave's 32-col N tile (0..3)

    const size_t bOff = (size_t)z * K * N;

    v8f acc00 = {}, acc01 = {}, acc10 = {}, acc11 = {};

    // ---- B-row elementwise transform (level-embed / GN+ReLU fusion)
    auto xform = [&](float v, int gk) -> float {
        if (kadd)   v += kadd[(z % kadd_mod) * K + gk];
        if (gnstat) {
            int g = gk / cpg;
            float mean = gnstat[(z * GG + g) * 2 + 0];
            float rstd = gnstat[(z * GG + g) * 2 + 1];
            v = (v - mean) * rstd * gamma[gk] + beta[gk];
        }
        if (relu) v = fmaxf(v, 0.f);
        return v;
    };

    // ---- staging: A tile 64x32 (branchless guard), B tile 32k x 128n
    auto stageAB = [&](int kt, int buf) {
        {
            int row = t >> 2;
            int kc  = (t & 3) * 8;
            int gm  = mBase + row;
            int gmc = gm < M ? gm : (M - 1);
            float msk = gm < M ? 1.f : 0.f;
            const float* src = &A[(size_t)gmc * K + (kt << 5) + kc];
            float4 f0 = *reinterpret_cast<const float4*>(src);
            float4 f1 = *reinterpret_cast<const float4*>(src + 4);
            __bf16* dst = &ldsA[buf][row * LDK + kc];
            dst[0] = (__bf16)(f0.x * msk); dst[1] = (__bf16)(f0.y * msk);
            dst[2] = (__bf16)(f0.z * msk); dst[3] = (__bf16)(f0.w * msk);
            dst[4] = (__bf16)(f1.x * msk); dst[5] = (__bf16)(f1.y * msk);
            dst[6] = (__bf16)(f1.z * msk); dst[7] = (__bf16)(f1.w * msk);
        }
        // two k-rows per thread -> packed b32 store into transposed [n][k] tile
#pragma unroll 2
        for (int i = 0; i < 8; ++i) {
            int idx = i * 256 + t;          // 0..2047
            int kk2 = (idx >> 7) << 1;      // even k within tile: 0,2,...,30
            int nn  = idx & 127;            // 0..127 (coalesced over t)
            int gk  = (kt << 5) + kk2;
            const float* bp = &Bm[bOff + (size_t)gk * N + (nBase + nn)];
            float v0 = xform(bp[0], gk);
            float v1 = xform(bp[N], gk + 1);
            unsigned pack = (unsigned)bf16bits(v0) |
                            ((unsigned)bf16bits(v1) << 16);
            *reinterpret_cast<unsigned*>(&ldsB[buf][nn * LDK + kk2]) = pack;
        }
    };

    stageAB(0, 0);
    __syncthreads();

    for (int kt = 0; kt < nk; ++kt) {
        if (kt + 1 < nk) stageAB(kt + 1, (kt + 1) & 1);
        if (kt + 2 < nk) {   // global_prefetch_b8 two tiles ahead
            __builtin_prefetch(
                &Bm[bOff + (size_t)(((kt + 2) << 5) + wid * 4) * N + nBase],
                0, 0);
        }

        const __bf16* sA = &ldsA[kt & 1][0];
        const __bf16* sB = &ldsB[kt & 1][0];

        // A frag: lane<16 -> K{0..7,16..23}; lane>=16 -> K{8..15,24..31}
        FragV fa0, fa1, fb0, fb1;
        int ra0 = (mw * 32 + r16) * LDK + hf * 8;
        int ra1 = ra0 + 16 * LDK;
        fa0.h[0] = *reinterpret_cast<const v8bf*>(&sA[ra0]);
        fa0.h[1] = *reinterpret_cast<const v8bf*>(&sA[ra0 + 16]);
        fa1.h[0] = *reinterpret_cast<const v8bf*>(&sA[ra1]);
        fa1.h[1] = *reinterpret_cast<const v8bf*>(&sA[ra1 + 16]);
        // B frag: lane<16 -> K{0..7,16..23}; lane>=16 -> K{8..15,24..31}
        int rb0 = (nw * 32 + r16) * LDK + hf * 8;
        int rb1 = rb0 + 16 * LDK;
        fb0.h[0] = *reinterpret_cast<const v8bf*>(&sB[rb0]);
        fb0.h[1] = *reinterpret_cast<const v8bf*>(&sB[rb0 + 16]);
        fb1.h[0] = *reinterpret_cast<const v8bf*>(&sB[rb1]);
        fb1.h[1] = *reinterpret_cast<const v8bf*>(&sB[rb1 + 16]);

        acc00 = __builtin_amdgcn_wmma_f32_16x16x32_bf16(
            false, fa0.v, false, fb0.v, (short)0, acc00, false, false);
        acc01 = __builtin_amdgcn_wmma_f32_16x16x32_bf16(
            false, fa0.v, false, fb1.v, (short)0, acc01, false, false);
        acc10 = __builtin_amdgcn_wmma_f32_16x16x32_bf16(
            false, fa1.v, false, fb0.v, (short)0, acc10, false, false);
        acc11 = __builtin_amdgcn_wmma_f32_16x16x32_bf16(
            false, fa1.v, false, fb1.v, (short)0, acc11, false, false);

        __syncthreads();
    }

    // ---- epilogue: C/D layout -> M = r + hf*8 (+16*mf), N = r16 (+16*nf)
    float scale = scale_ptr ? *scale_ptr : 0.f;
    auto store_tile = [&](const v8f& a, int mf, int nf) {
        int gmB  = mBase + mw * 32 + mf * 16 + hf * 8;
        int gcol = nBase + nw * 32 + nf * 16 + r16;
#pragma unroll
        for (int r = 0; r < 8; ++r) {
            int gm = gmB + r;
            if (gm >= M) break;
            float v = a[r];
            if (bias)    v += bias[gm];
            if (addfull) v += addfull[(size_t)z * M * N + (size_t)gm * N + gcol] * scale;
            if (out_transpose)
                Cm[(size_t)z * N * M + (size_t)gcol * M + gm] = v;
            else
                Cm[(size_t)z * M * N + (size_t)gm * N + gcol] = v;
        }
    };
    store_tile(acc00, 0, 0);
    store_tile(acc01, 0, 1);
    store_tile(acc10, 1, 0);
    store_tile(acc11, 1, 1);
}

// ---------------------------------------------------------------------------
// Softmax over the 12 (level,point) taps per (b, head, q); in-place.
// ---------------------------------------------------------------------------
__global__ __launch_bounds__(256) void softmax12_kernel(
    float* __restrict__ attn, int total)
{
    int idx = blockIdx.x * 256 + threadIdx.x;
    if (idx >= total) return;
    int q  = idx & (QQ - 1);
    int nh = (idx >> 12) & (NH - 1);
    int b  = idx >> 15;
    float* p = attn + ((size_t)(b * NH * 12 + nh * 12)) * QQ + q;
    float vals[12]; float mx = -1e30f;
    for (int j = 0; j < 12; ++j) { vals[j] = p[(size_t)j * QQ]; mx = fmaxf(mx, vals[j]); }
    float s = 0.f;
    for (int j = 0; j < 12; ++j) { vals[j] = __expf(vals[j] - mx); s += vals[j]; }
    float inv = 1.f / s;
    for (int j = 0; j < 12; ++j) p[(size_t)j * QQ] = vals[j] * inv;
}

// ---------------------------------------------------------------------------
// Bilinear sampling + attention-weighted sum.
// vproj is [b*L][q][c] (transposed GEMM output) -> lanes (d) read contiguous.
// ---------------------------------------------------------------------------
__global__ __launch_bounds__(256) void sample_kernel(
    const float* __restrict__ vproj, const float* __restrict__ attn,
    const float* __restrict__ offs,  float* __restrict__ outb)
{
    int b  = blockIdx.z;
    int nh = blockIdx.y;
    int q  = blockIdx.x * 8 + (threadIdx.x >> 5);
    int d  = threadIdx.x & 31;
    int qx = q & 63, qy = q >> 6;
    float rx = qx * (1.f / 63.f), ry = qy * (1.f / 63.f);
    int c = nh * DD + d;
    float acc = 0.f;
    for (int l = 0; l < LL; ++l) {
        const float* vp = vproj + ((size_t)(b * LL + l)) * QQ * CC + c;
        for (int p4 = 0; p4 < NP; ++p4) {
            int chA = nh * 12 + l * NP + p4;
            float w  = attn[((size_t)b * 96  + chA)         * QQ + q];
            float ox = offs[((size_t)b * 192 + chA * 2)     * QQ + q] * (1.f / 64.f);
            float oy = offs[((size_t)b * 192 + chA * 2 + 1) * QQ + q] * (1.f / 64.f);
            float sx = fminf(fmaxf(rx + ox, 0.f), 1.f) * 2.f - 1.f;
            float sy = fminf(fmaxf(ry + oy, 0.f), 1.f) * 2.f - 1.f;
            float ix = ((sx + 1.f) * 64.f - 1.f) * 0.5f;
            float iy = ((sy + 1.f) * 64.f - 1.f) * 0.5f;
            float x0f = floorf(ix), y0f = floorf(iy);
            float tx = ix - x0f, ty = iy - y0f;
            int x0 = (int)x0f, y0 = (int)y0f;
            float sval = 0.f;
#pragma unroll
            for (int tap = 0; tap < 4; ++tap) {
                int dx = tap & 1, dy = tap >> 1;
                int xc = x0 + dx, yc = y0 + dy;
                if (xc >= 0 && xc < 64 && yc >= 0 && yc < 64) {
                    float wt = (dx ? tx : 1.f - tx) * (dy ? ty : 1.f - ty);
                    sval += vp[(size_t)(yc * 64 + xc) * CC] * wt;
                }
            }
            acc += w * sval;
        }
    }
    outb[((size_t)b * CC + c) * QQ + q] = acc;
}

// ---------------------------------------------------------------------------
// Workspace layout (floats)
// ---------------------------------------------------------------------------
#define OFF_QLOW   ((size_t)0)           // 4*256*4096
#define OFF_QHIGH  ((size_t)4194304)
#define OFF_SOH    ((size_t)8388608)     // 4*128*4096
#define OFF_AWH    ((size_t)10485760)
#define OFF_GNSO   ((size_t)12582912)    // 64 (padded 256)
#define OFF_GNAW   ((size_t)12583168)
#define OFF_OFFS   ((size_t)12583424)    // 4*192*4096
#define OFF_ATTN   ((size_t)15729152)    // 4*96*4096
#define OFF_VPROJ  ((size_t)17302016)    // 12*4096*256
#define OFF_AOUT   ((size_t)29884928)    // 4*256*4096

extern "C" void kernel_launch(void* const* d_in, const int* in_sizes, int n_in,
                              void* d_out, int out_size, void* d_ws, size_t ws_size,
                              hipStream_t stream)
{
    (void)in_sizes; (void)n_in; (void)out_size; (void)ws_size;
    const float* query    = (const float*)d_in[0];
    // d_in[1] = keys (unused by reference)
    const float* values   = (const float*)d_in[2];
    const float* so_w1    = (const float*)d_in[3];
    const float* so_b1    = (const float*)d_in[4];
    const float* so_gn_g  = (const float*)d_in[5];
    const float* so_gn_b  = (const float*)d_in[6];
    const float* so_w2    = (const float*)d_in[7];
    const float* so_b2    = (const float*)d_in[8];
    const float* aw_w1    = (const float*)d_in[9];
    const float* aw_b1    = (const float*)d_in[10];
    const float* aw_gn_g  = (const float*)d_in[11];
    const float* aw_gn_b  = (const float*)d_in[12];
    const float* aw_w2    = (const float*)d_in[13];
    const float* aw_b2    = (const float*)d_in[14];
    const float* vp_w     = (const float*)d_in[15];
    const float* vp_b     = (const float*)d_in[16];
    const float* op_w     = (const float*)d_in[17];
    const float* op_b     = (const float*)d_in[18];
    const float* level_embed = (const float*)d_in[19];
    const float* hf_scale    = (const float*)d_in[20];

    float* ws    = (float*)d_ws;
    float* qlow  = ws + OFF_QLOW;
    float* qhigh = ws + OFF_QHIGH;
    float* soh   = ws + OFF_SOH;
    float* awh   = ws + OFF_AWH;
    float* gnso  = ws + OFF_GNSO;
    float* gnaw  = ws + OFF_GNAW;
    float* offsb = ws + OFF_OFFS;
    float* attnb = ws + OFF_ATTN;
    float* vproj = ws + OFF_VPROJ;
    float* aout  = ws + OFF_AOUT;
    float* outp  = (float*)d_out;

    // 1) frequency decoupling
    blur_kernel<<<dim3(16384), 256, 0, stream>>>(query, qlow, qhigh, BB * CC * QQ);

    // 2) MLP stage-1 GEMMs (C->CM) for offsets and attention branches
    wmma_gemm_bf16<<<dim3(QQ / 128, 2, BB), 256, 0, stream>>>(
        so_w1, qlow, soh, CM, QQ, CC, so_b1,
        nullptr, 1, nullptr, nullptr, nullptr, 16, 0, nullptr, nullptr, 0);
    wmma_gemm_bf16<<<dim3(QQ / 128, 2, BB), 256, 0, stream>>>(
        aw_w1, qlow, awh, CM, QQ, CC, aw_b1,
        nullptr, 1, nullptr, nullptr, nullptr, 16, 0, nullptr, nullptr, 0);

    // 3) GroupNorm stats (per batch, per group)
    gn_stats_kernel<<<dim3(GG, BB), 256, 0, stream>>>(soh, gnso, CM, QQ, CM / GG);
    gn_stats_kernel<<<dim3(GG, BB), 256, 0, stream>>>(awh, gnaw, CM, QQ, CM / GG);

    // 4) MLP stage-2 GEMMs with fused GN+ReLU on the B operand
    wmma_gemm_bf16<<<dim3(QQ / 128, 3, BB), 256, 0, stream>>>(
        so_w2, soh, offsb, NH * LL * NP * 2, QQ, CM, so_b2,
        nullptr, 1, gnso, so_gn_g, so_gn_b, CM / GG, 1, nullptr, nullptr, 0);
    wmma_gemm_bf16<<<dim3(QQ / 128, 2, BB), 256, 0, stream>>>(
        aw_w2, awh, attnb, NH * LL * NP, QQ, CM, aw_b2,
        nullptr, 1, gnaw, aw_gn_g, aw_gn_b, CM / GG, 1, nullptr, nullptr, 0);

    // 5) softmax over 12 taps per (b, head, q)
    softmax12_kernel<<<dim3((BB * NH * QQ) / 256), 256, 0, stream>>>(
        attnb, BB * NH * QQ);

    // 6) value projection (dominant GEMM), level-embed fused, transposed out
    wmma_gemm_bf16<<<dim3(QQ / 128, 4, BB * LL), 256, 0, stream>>>(
        vp_w, values, vproj, CC, QQ, CC, vp_b,
        level_embed, LL, nullptr, nullptr, nullptr, 16, 0, nullptr, nullptr, 1);

    // 7) bilinear sampling + weighted sum
    sample_kernel<<<dim3(QQ / 8, NH, BB), 256, 0, stream>>>(
        vproj, attnb, offsb, aout);

    // 8) output projection + q_high * hf_scale residual, straight to d_out
    wmma_gemm_bf16<<<dim3(QQ / 128, 4, BB), 256, 0, stream>>>(
        op_w, aout, outp, CC, QQ, CC, op_b,
        nullptr, 1, nullptr, nullptr, nullptr, 16, 0, qhigh, hf_scale, 0);
}
